// CharRNN_86792699118072
// MI455X (gfx1250) — matrix operand order repaired and checked by reading, not compiled, not run on hardware
//
#include <hip/hip_runtime.h>
#include <hip/hip_bf16.h>

// Problem dims
#define B_ 256
#define L_ 512
#define E_ 256
#define H_ 1024
#define V_ 64
#define NWG 64        // persistent workgroups (1 per WGP)
#define PADK 1032     // LDS row stride in bf16 elems: 516 dwords == 4 mod 64 banks

typedef __bf16    v16bf __attribute__((ext_vector_type(16)));
typedef float     v8f   __attribute__((ext_vector_type(8)));
typedef unsigned  u32x4 __attribute__((ext_vector_type(4)));
typedef int       i32x4 __attribute__((ext_vector_type(4)));

// Address-space qualified vector types for the async global->LDS builtin
typedef __attribute__((address_space(1))) i32x4 gi32x4;
typedef __attribute__((address_space(3))) i32x4 li32x4;

union ABfrag { v16bf v; u32x4 q[2]; };  // 32 bytes = 8 VGPRs

__device__ __forceinline__ unsigned short f2bf(float f) {
  unsigned u = __builtin_bit_cast(unsigned, f);
  u += 0x7fffu + ((u >> 16) & 1u);      // round-to-nearest-even
  return (unsigned short)(u >> 16);
}

// ---------------------------------------------------------------------------
// Async global -> LDS copy (gfx1250 ASYNCcnt path), guarded fallback
// ---------------------------------------------------------------------------
__device__ __forceinline__ void cp16_async(const unsigned short* g, unsigned short* l) {
#if __has_builtin(__builtin_amdgcn_global_load_async_to_lds_b128)
  __builtin_amdgcn_global_load_async_to_lds_b128((gi32x4*)g, (li32x4*)l, 0, 0);
#else
  *(u32x4*)l = *(const u32x4*)g;
#endif
}

__device__ __forceinline__ void async_wait0() {
#if __has_builtin(__builtin_amdgcn_s_wait_asynccnt)
  __builtin_amdgcn_s_wait_asynccnt(0);
#elif __has_builtin(__builtin_amdgcn_global_load_async_to_lds_b128)
  asm volatile("s_wait_asynccnt 0x0" ::: "memory");
#endif
}

// Stage nrows x 1024 bf16 (row stride H_ in global) into LDS with PADK stride.
__device__ __forceinline__ void stage_rows(const unsigned short* __restrict__ g,
                                           unsigned short* l, int nrows, int tid) {
  int total = nrows * (H_ / 8);              // 16-byte chunks
  for (int c = tid; c < total; c += 256) {
    int r = c >> 7;
    int col = (c & 127) * 8;
    cp16_async(g + r * H_ + col, l + r * PADK + col);
  }
}

// ---------------------------------------------------------------------------
// Prep kernels
// ---------------------------------------------------------------------------
__global__ void prep_init(unsigned* __restrict__ h0_u32, unsigned* __restrict__ bar) {
  int tid = blockIdx.x * 256 + threadIdx.x;
  if (tid < (B_ * H_ * 2 / 4)) h0_u32[tid] = 0u;
  if (tid == 0) { bar[0] = 0u; bar[1] = 0u; }
}

// proj_table[v][h] = dot(embedding[v,:], W_ih[:,h]) + b_h[h]   (64 x 1024 f32)
__global__ void prep_table(const float* __restrict__ emb, const float* __restrict__ wih,
                           const float* __restrict__ bh, float* __restrict__ proj) {
  int tid = blockIdx.x * 256 + threadIdx.x;    // 65536 total
  int v = tid >> 10, h = tid & (H_ - 1);
  const float* e = emb + v * E_;
  float s = bh[h];
  for (int k = 0; k < E_; ++k) s += e[k] * wih[k * H_ + h];
  proj[tid] = s;
}

// whhT[n][k] = bf16(W_hh[k][n]);  whoT[v][k] = bf16(W_ho[k][v])
__global__ void prep_transpose(const float* __restrict__ whh, const float* __restrict__ who,
                               unsigned short* __restrict__ whhT,
                               unsigned short* __restrict__ whoT) {
  int tid = blockIdx.x * 256 + threadIdx.x;
  if (tid < H_ * H_) {
    int n = tid >> 10, k = tid & (H_ - 1);
    whhT[tid] = f2bf(whh[k * H_ + n]);
  } else if (tid < H_ * H_ + V_ * H_) {
    int i = tid - H_ * H_;
    int v = i >> 10, k = i & (H_ - 1);
    whoT[i] = f2bf(who[k * V_ + v]);
  }
}

// ---------------------------------------------------------------------------
// Grid-wide barrier (phase counter reset each call by prep_init)
// ---------------------------------------------------------------------------
__device__ __forceinline__ void grid_barrier(unsigned* cnt, unsigned* phase,
                                             unsigned target) {
  __threadfence();
  __syncthreads();
  if (threadIdx.x == 0) {
    unsigned prev = __hip_atomic_fetch_add(cnt, 1u, __ATOMIC_ACQ_REL,
                                           __HIP_MEMORY_SCOPE_AGENT);
    if (prev == NWG - 1u) {
      __hip_atomic_store(cnt, 0u, __ATOMIC_RELAXED, __HIP_MEMORY_SCOPE_AGENT);
      __hip_atomic_fetch_add(phase, 1u, __ATOMIC_RELEASE, __HIP_MEMORY_SCOPE_AGENT);
    } else {
      while (__hip_atomic_load(phase, __ATOMIC_ACQUIRE,
                               __HIP_MEMORY_SCOPE_AGENT) < target) {
        __builtin_amdgcn_s_sleep(1);
      }
    }
  }
  __syncthreads();
}

// ---------------------------------------------------------------------------
// Persistent RNN kernel: 64 blocks x 256 threads (8 wave32)
//   LDS: Astage 64x1024 (h_{t-1}, restaged per step, async engine)
//        Bstage 64x1024 (block's W_hh^T cols, staged once)
//        Wstage 16x1024 (block's W_ho^T cols, staged once)   total ~290 KB
//   h-step : wave w -> 16 batch rows x 32 hidden cols (two 16x16x32 bf16 WMMA accs)
//   logits : wave 0 -> 16 rows x 16 vocab for step t-1, entirely from LDS
// ---------------------------------------------------------------------------
__global__ void __launch_bounds__(256, 1)
rnn_persistent(const int* __restrict__ x, const float* __restrict__ proj,
               const unsigned short* __restrict__ whhT,
               const unsigned short* __restrict__ whoT,
               const float* __restrict__ bo,
               unsigned short* __restrict__ hb,   // 2*B_*H_ bf16 ping-pong (first half zeroed)
               unsigned* __restrict__ bar, float* __restrict__ out) {
  extern __shared__ char smem[];
  unsigned short* Ast = (unsigned short*)smem;                       // 64 x PADK
  unsigned short* Bst = (unsigned short*)(smem + 64 * PADK * 2);     // 64 x PADK
  unsigned short* Wst = (unsigned short*)(smem + 128 * PADK * 2);    // 16 x PADK

  const int tid  = threadIdx.x;
  const int wave = tid >> 5;
  const int lane = tid & 31;
  const int lh   = lane & 15;
  const int hi   = lane >> 4;
  const int blk  = blockIdx.x;
  const int br   = blk >> 4;                 // batch block 0..3  (64 rows)
  const int bc   = blk & 15;                 // hidden block 0..15 (64 cols)

  const int growbase = br * 64;              // global batch row base
  const int gcolbase = bc * 64;              // global hidden col base
  const int lrowA = (wave >> 1) * 16;        // wave's local A row base
  const int lcolB = (wave & 1) * 32;         // wave's local B col base

  // logits mapping: vocab cols (bc>>2)*16.., batch rows growbase + (bc&3)*16..
  const int vcol  = (bc >> 2) * 16;
  const int lrowL = (bc & 3) * 16;
  const float bo_val = bo[vcol + lh];

  // stage time-invariant weights into LDS once
  stage_rows(whhT + gcolbase * H_, Bst, 64, tid);
  stage_rows(whoT + vcol * H_,     Wst, 16, tid);
  async_wait0();
  __syncthreads();

  const int FINAL_OFF = B_ * L_ * V_;

#pragma unroll 1
  for (int t = 0; t <= L_; ++t) {
    // stage A <- h_{t-1} for this block's 64 batch rows (async engine)
    const unsigned short* hp = hb + (t & 1) * (B_ * H_);
    stage_rows(hp + growbase * H_, Ast, 64, tid);
    async_wait0();
    __syncthreads();

    // ---- fused output projection for step t-1 (pure LDS) ----
    if (wave == 0 && t >= 1) {
      v8f lacc;
#pragma unroll
      for (int i = 0; i < 8; ++i) lacc[i] = bo_val;
      const unsigned short* la = Ast + (lrowL + lh) * PADK;
      const unsigned short* lb = Wst + lh * PADK;
      for (int k0 = 0; k0 < H_; k0 += 32) {
        ABfrag a, b;
        a.q[0] = *(const u32x4*)(la + k0 + hi * 16);
        a.q[1] = *(const u32x4*)(la + k0 + hi * 16 + 8);
        const unsigned short* p = lb + k0 + hi * 16;
        b.q[0] = *(const u32x4*)(p);
        b.q[1] = *(const u32x4*)(p + 8);
        lacc = __builtin_amdgcn_wmma_f32_16x16x32_bf16(false, a.v, false, b.v,
                                                       (short)0, lacc, false, false);
      }
#pragma unroll
      for (int i = 0; i < 8; ++i) {
        __builtin_nontemporal_store(
            lacc[i], out + (growbase + lrowL + i + 8 * hi) * (L_ * V_) +
                         (t - 1) * V_ + vcol + lh);
      }
    }

    if (t < L_) {
      unsigned short* hc = hb + ((t + 1) & 1) * (B_ * H_);   // h_t

      // ---- accumulator init: x_proj gather ----
      v8f acc0, acc1;
#pragma unroll
      for (int i = 0; i < 8; ++i) {
        int m = growbase + lrowA + i + hi * 8;
        int xi = x[m * L_ + t];
        acc0[i] = proj[xi * H_ + gcolbase + lcolB + lh];
        acc1[i] = proj[xi * H_ + gcolbase + lcolB + 16 + lh];
      }

      // ---- h_prev @ W_hh from LDS via v_wmma_f32_16x16x32_bf16 ----
      const unsigned short* la  = Ast + (lrowA + lh) * PADK;
      const unsigned short* lb0 = Bst + (lcolB + lh) * PADK;
      const unsigned short* lb1 = lb0 + 16 * PADK;
      for (int k0 = 0; k0 < H_; k0 += 32) {
        ABfrag a, b0, b1;
        a.q[0] = *(const u32x4*)(la + k0 + hi * 16);
        a.q[1] = *(const u32x4*)(la + k0 + hi * 16 + 8);
        const unsigned short* p0 = lb0 + k0 + hi * 16;
        b0.q[0] = *(const u32x4*)(p0);
        b0.q[1] = *(const u32x4*)(p0 + 8);
        const unsigned short* p1 = lb1 + k0 + hi * 16;
        b1.q[0] = *(const u32x4*)(p1);
        b1.q[1] = *(const u32x4*)(p1 + 8);
        acc0 = __builtin_amdgcn_wmma_f32_16x16x32_bf16(false, a.v, false, b0.v,
                                                       (short)0, acc0, false, false);
        acc1 = __builtin_amdgcn_wmma_f32_16x16x32_bf16(false, a.v, false, b1.v,
                                                       (short)0, acc1, false, false);
      }

      // ---- tanh, publish h_t (bf16); final hidden in f32 at t = L-1 ----
#pragma unroll
      for (int i = 0; i < 8; ++i) {
        int m = growbase + lrowA + i + hi * 8;
        float t0 = tanhf(acc0[i]);
        float t1 = tanhf(acc1[i]);
        hc[m * H_ + gcolbase + lcolB + lh]      = f2bf(t0);
        hc[m * H_ + gcolbase + lcolB + 16 + lh] = f2bf(t1);
        if (t == L_ - 1) {
          __builtin_nontemporal_store(t0, out + FINAL_OFF + m * H_ + gcolbase + lcolB + lh);
          __builtin_nontemporal_store(t1, out + FINAL_OFF + m * H_ + gcolbase + lcolB + 16 + lh);
        }
      }

      grid_barrier(&bar[0], &bar[1], (unsigned)(t + 1));
    }
  }
}

// ---------------------------------------------------------------------------
extern "C" void kernel_launch(void* const* d_in, const int* in_sizes, int n_in,
                              void* d_out, int out_size, void* d_ws, size_t ws_size,
                              hipStream_t stream) {
  const int*   x   = (const int*)  d_in[0];
  const float* emb = (const float*)d_in[1];
  const float* wih = (const float*)d_in[2];
  const float* whh = (const float*)d_in[3];
  const float* bh  = (const float*)d_in[4];
  const float* who = (const float*)d_in[5];
  const float* bo  = (const float*)d_in[6];
  float* out = (float*)d_out;

  // Workspace layout (~3.4 MB; fully L2-resident, L2 = 192 MB)
  char* ws = (char*)d_ws;
  unsigned*       bar  = (unsigned*)ws;                               // 256 B
  float*          proj = (float*)(ws + 256);                          // 256 KB
  unsigned short* whhT = (unsigned short*)(ws + 256 + V_ * H_ * 4);   // 2 MB
  unsigned short* whoT = (unsigned short*)((char*)whhT + H_ * H_ * 2);// 128 KB
  unsigned short* hb   = (unsigned short*)((char*)whoT + V_ * H_ * 2);// 2 x 512 KB ping-pong

  const size_t smem_bytes = (size_t)(128 + 16) * PADK * 2;            // ~290 KB

  prep_init<<<513, 256, 0, stream>>>((unsigned*)hb, bar);
  prep_table<<<(V_ * H_) / 256, 256, 0, stream>>>(emb, wih, bh, proj);
  prep_transpose<<<(H_ * H_ + V_ * H_) / 256, 256, 0, stream>>>(whh, who, whhT, whoT);
  rnn_persistent<<<NWG, 256, smem_bytes, stream>>>(x, proj, whhT, whoT, bo, hb, bar, out);
}
